// MultiHeadSelfAttention_54674933678805
// MI455X (gfx1250) — matrix operand order repaired and checked by reading, not compiled, not run on hardware
//
#include <hip/hip_runtime.h>
#include <hip/hip_bf16.h>

typedef __attribute__((ext_vector_type(16))) _Float16 v16h;
typedef __attribute__((ext_vector_type(8)))  _Float16 v8h;
typedef __attribute__((ext_vector_type(8)))  float    v8f;
typedef __attribute__((ext_vector_type(4)))  int      v4i;

// address-space-qualified element types for the async-to-LDS builtin:
// param0 = pointer to int4 in global (AS1), param1 = pointer to int4 in LDS (AS3)
typedef __attribute__((address_space(1))) v4i as1_v4i;
typedef __attribute__((address_space(3))) v4i as3_v4i;

#define WMMA_F32_F16(a, b, c) \
    __builtin_amdgcn_wmma_f32_16x16x32_f16(false, (a), false, (b), (short)0, (c), false, false)

#if __has_builtin(__builtin_amdgcn_global_load_async_to_lds_b128) && \
    __has_builtin(__builtin_amdgcn_s_wait_asynccnt)
#define HAS_ASYNC_LDS 1
#endif

// ---------------------------------------------------------------- constants
#define BATCH   8
#define SEQ     1024
#define EMBED   768
#define HEADS   12
#define HDIM    64
#define THREE_D 2304

#define BK       32          // K-step per WMMA sweep
#define BN       64          // N tile per block
#define BSTRIDE  72          // padded LDS row stride (halfs): 144B, 16B aligned

// ---------------------------------------------------------------- helpers
// Stage one 32x64 f16 B tile (wT[kb..kb+31][nbase..nbase+63]) into LDS.
// 256 threads x 16B = 4KB = whole tile; async when available.
__device__ __forceinline__ void stage_b_tile(const _Float16* __restrict__ wT,
                                             int ldw, int kb, int nbase,
                                             _Float16* lds_tile, int tid) {
    int r = tid >> 3;            // 0..31  (K row)
    int c = (tid & 7) * 8;       // 0..56  (N col, 8-half chunks)
    const _Float16* g = wT + (size_t)(kb + r) * ldw + nbase + c;
    _Float16* l = lds_tile + r * BSTRIDE + c;
#ifdef HAS_ASYNC_LDS
    __builtin_amdgcn_global_load_async_to_lds_b128(
        (as1_v4i*)g, (as3_v4i*)l, 0, 0);
#else
    *(v8h*)l = *(const v8h*)g;
#endif
}

__device__ __forceinline__ v16h load_b_frag(const _Float16* lds_tile,
                                            int lane, int nt) {
    const _Float16* bp = lds_tile + (lane >> 4) * 8 * BSTRIDE + nt * 16 + (lane & 15);
    v16h bf;
#pragma unroll
    for (int i = 0; i < 8; ++i) { bf[i] = bp[i * BSTRIDE]; bf[8 + i] = bp[(16 + i) * BSTRIDE]; }
    return bf;
}

// A fragment: lane holds row `rowp`, K = kb + khalf*8 + {0..7, 16..23}
__device__ __forceinline__ v16h load_a_frag(const _Float16* rowp,
                                            int kb, int khalf) {
    const _Float16* ap = rowp + kb + khalf * 8;
    v8h a0 = *(const v8h*)ap;
    v8h a1 = *(const v8h*)(ap + 16);
    v16h a;
#pragma unroll
    for (int i = 0; i < 8; ++i) { a[i] = a0[i]; a[8 + i] = a1[i]; }
    return a;
}

// ---------------------------------------------------------------- converters
__global__ void k_cvt_f32_f16(const float* __restrict__ in,
                              _Float16* __restrict__ out, int n) {
    int i = blockIdx.x * 256 + threadIdx.x;
    if (i < n) out[i] = (_Float16)in[i];
}

// out[c*R + r] = in[r*C + c]  (in: [R][C] f32 -> out: [C][R] f16)
__global__ void k_tcvt_f32_f16(const float* __restrict__ in,
                               _Float16* __restrict__ out, int R, int C) {
    int i = blockIdx.x * 256 + threadIdx.x;
    if (i < R * C) {
        int r = i / C, c = i - r * C;
        out[(size_t)c * R + r] = (_Float16)in[i];
    }
}

// ---------------------------------------------------------------- QKV GEMM
// qkv[m][c] = sum_k xh[m][k] * wT[k][c] + bias[c]
// scatter: t==0 -> Q[b,h,nq,hd]; t==1 -> Kt[b,h,hd,nq]; t==2 -> V[b,h,nq,hd]
__global__ void k_qkv_gemm(const _Float16* __restrict__ xh,    // [8192][768]
                           const _Float16* __restrict__ wT,    // [768][2304]
                           const float*    __restrict__ bias,  // [2304]
                           _Float16* __restrict__ Q,
                           _Float16* __restrict__ Kt,
                           _Float16* __restrict__ V) {
    __shared__ __align__(16) _Float16 btile[2][BK * BSTRIDE];

    const int tid   = threadIdx.x;
    const int lane  = tid & 31;
    const int w     = tid >> 5;
    const int mbase = blockIdx.x * 128 + w * 16;
    const int nbase = blockIdx.y * BN;
    const int l16   = lane & 15;
    const int khalf = lane >> 4;
    const _Float16* arow = xh + (size_t)(mbase + l16) * EMBED;

    stage_b_tile(wT, THREE_D, 0, nbase, btile[0], tid);
    v16h a = load_a_frag(arow, 0, khalf);

    v8f acc[4] = {};
    for (int kb = 0; kb < EMBED; kb += BK) {
        const int buf = (kb >> 5) & 1;
        const int kn  = (kb + BK < EMBED) ? (kb + BK) : kb;   // clamp: last step refetches
        stage_b_tile(wT, THREE_D, kn, nbase, btile[buf ^ 1], tid);
#ifdef HAS_ASYNC_LDS
        __builtin_amdgcn_s_wait_asynccnt(1);   // current tile done, next in flight
#endif
        __syncthreads();

        v16h anext = load_a_frag(arow, kn, khalf);   // pipeline A one step ahead

        v16h bf[4];
#pragma unroll
        for (int nt = 0; nt < 4; ++nt) bf[nt] = load_b_frag(btile[buf], lane, nt);
#pragma unroll
        for (int nt = 0; nt < 4; ++nt) acc[nt] = WMMA_F32_F16(a, bf[nt], acc[nt]);

        __syncthreads();   // all waves done reading before buf is overwritten
        a = anext;
    }

    // epilogue: D layout -> lane holds col n = l16, rows m = v + 8*khalf
#pragma unroll
    for (int nt = 0; nt < 4; ++nt) {
        int c  = nbase + nt * 16 + l16;
        int t  = c / EMBED;          // uniform per 16-wide tile
        int d  = c - t * EMBED;
        int h  = d >> 6;             // uniform per tile
        int hd = d & 63;
        float bv = bias[c];
#pragma unroll
        for (int v = 0; v < 8; ++v) {
            int m  = mbase + v + 8 * khalf;
            int bb = m >> 10;
            int nq = m & 1023;
            _Float16 hval = (_Float16)(acc[nt][v] + bv);
            size_t bh = (size_t)(bb * HEADS + h);
            if (t == 0)      Q [(bh * SEQ + nq) * HDIM + hd] = hval;
            else if (t == 1) Kt[(bh * HDIM + hd) * SEQ + nq] = hval;
            else             V [(bh * SEQ + nq) * HDIM + hd] = hval;
        }
    }
}

// ---------------------------------------------------------------- attention
// One wave handles 16 query rows of one (b,h); flash-style online softmax.
__global__ void k_attn(const _Float16* __restrict__ Q,
                       const _Float16* __restrict__ Kt,
                       const _Float16* __restrict__ V,
                       _Float16* __restrict__ ah) {   // [8192][768] merged heads
    __shared__ __align__(32) _Float16 psh[4][16 * 32];

    const int lane  = threadIdx.x & 31;
    const int w     = threadIdx.x >> 5;
    const int l16   = lane & 15;
    const int khalf = lane >> 4;

    int g  = blockIdx.x * 4 + w;                 // [0, 6144)
    int b  = g / (HEADS * 64);
    int r  = g - b * (HEADS * 64);
    int h  = r >> 6;
    int qbase = (r & 63) * 16;

    const size_t bh = (size_t)(b * HEADS + h);
    const _Float16* Qp  = Q  + (bh * SEQ + qbase + l16) * HDIM;
    const _Float16* Ktp = Kt + bh * HDIM * SEQ;
    const _Float16* Vp  = V  + bh * SEQ * HDIM;

    // Q A-fragments for hd 0..31 and 32..63
    v16h aq[2];
#pragma unroll
    for (int kk = 0; kk < 2; ++kk) aq[kk] = load_a_frag(Qp, kk * 32, khalf);

    float mi[8], li[8];
#pragma unroll
    for (int v = 0; v < 8; ++v) { mi[v] = -1e30f; li[v] = 0.f; }
    v8f O[4] = {};

    for (int kb = 0; kb < SEQ; kb += 32) {
        // S = Q @ K^T over hd=64 (two K-steps); two 16-col tiles
        v8f s0 = {}, s1 = {};
#pragma unroll
        for (int kk = 0; kk < 2; ++kk) {
            const _Float16* kp = Ktp + (size_t)(kk * 32 + khalf * 8) * SEQ + kb + l16;
            v16h b0, b1;
#pragma unroll
            for (int i = 0; i < 8; ++i) {
                b0[i] = kp[(size_t)i * SEQ];      b0[8 + i] = kp[(size_t)(16 + i) * SEQ];
                b1[i] = kp[(size_t)i * SEQ + 16]; b1[8 + i] = kp[(size_t)(16 + i) * SEQ + 16];
            }
            s0 = WMMA_F32_F16(aq[kk], b0, s0);
            s1 = WMMA_F32_F16(aq[kk], b1, s1);
        }
        // online softmax per row (row = v + 8*khalf, cols spread over 16 lanes)
#pragma unroll
        for (int v = 0; v < 8; ++v) {
            float sm0 = s0[v] * 0.125f;         // Hd^-0.5
            float sm1 = s1[v] * 0.125f;
            float t = fmaxf(sm0, sm1);
#pragma unroll
            for (int off = 1; off < 16; off <<= 1)
                t = fmaxf(t, __shfl_xor(t, off, 32));
            float nm = fmaxf(mi[v], t);
            float sc = __expf(mi[v] - nm);
            float p0 = __expf(sm0 - nm);
            float p1 = __expf(sm1 - nm);
            float rs = p0 + p1;
#pragma unroll
            for (int off = 1; off < 16; off <<= 1)
                rs += __shfl_xor(rs, off, 32);
            li[v] = li[v] * sc + rs;
            mi[v] = nm;
#pragma unroll
            for (int j = 0; j < 4; ++j) O[j][v] *= sc;
            int row = v + 8 * khalf;
            psh[w][row * 32 + l16]      = (_Float16)p0;
            psh[w][row * 32 + 16 + l16] = (_Float16)p1;
        }
        __syncthreads();   // uniform; relayout P (D-layout -> A-layout) via LDS
        v16h pa;
        {
            const _Float16* pp = &psh[w][l16 * 32 + khalf * 8];
            v8h lo = *(const v8h*)pp;
            v8h hi = *(const v8h*)(pp + 16);
#pragma unroll
            for (int i = 0; i < 8; ++i) { pa[i] = lo[i]; pa[8 + i] = hi[i]; }
        }
        // O += P @ V ; V B-fragment: lane holds hd column j*16+l16, keys kb + khalf*8 + {0..7, 16..23}
#pragma unroll
        for (int j = 0; j < 4; ++j) {
            const _Float16* vp = Vp + (size_t)(kb + khalf * 8) * HDIM + j * 16 + l16;
            v16h bv;
#pragma unroll
            for (int i = 0; i < 8; ++i) { bv[i] = vp[i * HDIM]; bv[8 + i] = vp[(16 + i) * HDIM]; }
            O[j] = WMMA_F32_F16(pa, bv, O[j]);
        }
        __syncthreads();
    }
    // normalize and store merged-head f16 output
#pragma unroll
    for (int v = 0; v < 8; ++v) {
        float inv = 1.0f / li[v];
        int row = v + 8 * khalf;
        size_t obase = ((size_t)(b * SEQ) + qbase + row) * EMBED + h * HDIM + l16;
#pragma unroll
        for (int j = 0; j < 4; ++j)
            ah[obase + j * 16] = (_Float16)(O[j][v] * inv);
    }
}

// ---------------------------------------------------------------- proj GEMM
__global__ void k_proj_gemm(const _Float16* __restrict__ ahh,   // [8192][768]
                            const _Float16* __restrict__ wT,    // [768][768]
                            const float*    __restrict__ bias,  // [768]
                            float* __restrict__ out) {          // [8192][768]
    __shared__ __align__(16) _Float16 btile[2][BK * BSTRIDE];

    const int tid   = threadIdx.x;
    const int lane  = tid & 31;
    const int w     = tid >> 5;
    const int mbase = blockIdx.x * 128 + w * 16;
    const int nbase = blockIdx.y * BN;
    const int l16   = lane & 15;
    const int khalf = lane >> 4;
    const _Float16* arow = ahh + (size_t)(mbase + l16) * EMBED;

    stage_b_tile(wT, EMBED, 0, nbase, btile[0], tid);
    v16h a = load_a_frag(arow, 0, khalf);

    v8f acc[4] = {};
    for (int kb = 0; kb < EMBED; kb += BK) {
        const int buf = (kb >> 5) & 1;
        const int kn  = (kb + BK < EMBED) ? (kb + BK) : kb;
        stage_b_tile(wT, EMBED, kn, nbase, btile[buf ^ 1], tid);
#ifdef HAS_ASYNC_LDS
        __builtin_amdgcn_s_wait_asynccnt(1);
#endif
        __syncthreads();

        v16h anext = load_a_frag(arow, kn, khalf);

        v16h bf[4];
#pragma unroll
        for (int nt = 0; nt < 4; ++nt) bf[nt] = load_b_frag(btile[buf], lane, nt);
#pragma unroll
        for (int nt = 0; nt < 4; ++nt) acc[nt] = WMMA_F32_F16(a, bf[nt], acc[nt]);

        __syncthreads();
        a = anext;
    }
#pragma unroll
    for (int nt = 0; nt < 4; ++nt) {
        int c = nbase + nt * 16 + l16;
        float bv = bias[c];
#pragma unroll
        for (int v = 0; v < 8; ++v) {
            int m = mbase + v + 8 * khalf;
            out[(size_t)m * EMBED + c] = acc[nt][v] + bv;
        }
    }
}

// ---------------------------------------------------------------- launch
extern "C" void kernel_launch(void* const* d_in, const int* in_sizes, int n_in,
                              void* d_out, int out_size, void* d_ws, size_t ws_size,
                              hipStream_t stream) {
    const float* x     = (const float*)d_in[0];   // [8,1024,768]
    const float* Wqkv  = (const float*)d_in[1];   // [2304,768]
    const float* bqkv  = (const float*)d_in[2];   // [2304]
    const float* Wproj = (const float*)d_in[3];   // [768,768]
    const float* bproj = (const float*)d_in[4];   // [768]
    float* out = (float*)d_out;

    char* p = (char*)d_ws;
    _Float16* xh     = (_Float16*)p; p += (size_t)BATCH * SEQ * EMBED * 2;
    _Float16* wqkvT  = (_Float16*)p; p += (size_t)EMBED * THREE_D * 2;
    _Float16* wprojT = (_Float16*)p; p += (size_t)EMBED * EMBED * 2;
    _Float16* Q      = (_Float16*)p; p += (size_t)BATCH * HEADS * SEQ * HDIM * 2;
    _Float16* Kt     = (_Float16*)p; p += (size_t)BATCH * HEADS * SEQ * HDIM * 2;
    _Float16* V      = (_Float16*)p; p += (size_t)BATCH * HEADS * SEQ * HDIM * 2;
    _Float16* ah     = (_Float16*)p; p += (size_t)BATCH * SEQ * EMBED * 2;

    const int nx = BATCH * SEQ * EMBED;
    k_cvt_f32_f16<<<(nx + 255) / 256, 256, 0, stream>>>(x, xh, nx);
    k_tcvt_f32_f16<<<(THREE_D * EMBED + 255) / 256, 256, 0, stream>>>(Wqkv, wqkvT, THREE_D, EMBED);
    k_tcvt_f32_f16<<<(EMBED * EMBED + 255) / 256, 256, 0, stream>>>(Wproj, wprojT, EMBED, EMBED);

    k_qkv_gemm<<<dim3(BATCH * SEQ / 128, THREE_D / BN), 256, 0, stream>>>(
        xh, wqkvT, bqkv, Q, Kt, V);

    k_attn<<<BATCH * HEADS * (SEQ / 16) / 4, 128, 0, stream>>>(Q, Kt, V, ah);

    k_proj_gemm<<<dim3(BATCH * SEQ / 128, EMBED / BN), 256, 0, stream>>>(
        ah, wprojT, bproj, out);
}
